// TransformerBlock_74594991997052
// MI455X (gfx1250) — matrix operand order repaired and checked by reading, not compiled, not run on hardware
//
#include <hip/hip_runtime.h>
#include <hip/hip_bf16.h>

// ---------------- problem constants ----------------
constexpr int cB  = 4;
constexpr int cS  = 1024;
constexpr int cD  = 512;
constexpr int cH  = 8;
constexpr int cDK = 64;
constexpr int cFF = 2048;
constexpr int cNB = 65;          // 2*MRP+1 relative-position buckets
constexpr float cSCALE = 0.125f; // 1/sqrt(64)

typedef __bf16 bf16;
typedef __attribute__((ext_vector_type(16))) __bf16 v16bf;
typedef __attribute__((ext_vector_type(8)))  __bf16 v8bf;
typedef __attribute__((ext_vector_type(8)))  float  v8f;
typedef __attribute__((ext_vector_type(4)))  unsigned int u32x4;
typedef __attribute__((ext_vector_type(8)))  int i32x8;
typedef __attribute__((ext_vector_type(4)))  int i32x4;

union V16 { v16bf v; v8bf h[2]; bf16 e[16]; };
union C8  { v8f  v; float f[8]; };

// ---------------- WMMA fragment loads (gfx1250 layouts, vectorized) ----------------
// A fragment 16x32 bf16 from row-major source, leading dim ld (elements).
// Per ISA 7.12.2: lane m=lane&15 holds row M=m; per lane the K values are two
// contiguous 8-element chunks: K = 8*kh+0..7 (VGPR0-3) and 16+8*kh+0..7 (VGPR4-7).
// -> two b128 loads per lane. Requires row base 16B-aligned (ld multiple of 8).
static __device__ inline v16bf load_a_frag(const bf16* __restrict__ p, int ld, int lane) {
  int m  = lane & 15;
  int kh = (lane >> 4) & 1;
  const bf16* row = p + (size_t)m * ld + kh * 8;
  V16 f;
  f.h[0] = *(const v8bf*)(row);
  f.h[1] = *(const v8bf*)(row + 16);
  return f.v;
}

// B fragment 32x16 bf16 where the SOURCE is stored transposed: element (k,n) at
// p[n*ld + k]. Per lane (col n=lane&15): K = 8*kh+0..7 and 16+8*kh+0..7
// -> two b128 loads per lane.
static __device__ inline v16bf load_bT_frag(const bf16* __restrict__ p, int ld, int lane) {
  int n  = lane & 15;
  int kh = (lane >> 4) & 1;
  const bf16* row = p + (size_t)n * ld + kh * 8;
  V16 f;
  f.h[0] = *(const v8bf*)(row);
  f.h[1] = *(const v8bf*)(row + 16);
  return f.v;
}

static __device__ inline v8f wmma_bf16(v16bf a, v16bf b, v8f c) {
  return __builtin_amdgcn_wmma_f32_16x16x32_bf16(false, a, false, b, (short)0, c, false, false);
}

static __device__ inline int iclamp(int x, int lo, int hi) {
  return x < lo ? lo : (x > hi ? hi : x);
}

// ---------------- TDM: 2D bf16 tile -> LDS (D# built per CDNA5 ISA) ----------------
// Loads tile_d1 rows x tile_d0 cols (bf16) from row-major tensor with row stride
// stride_elems into LDS at lds_off. Optional padding: pad_amount+1 DWORDs inserted
// after every 2<<pad_interval DWORDs (0/0 = no padding).
// NOTE: this toolchain exposes the 6-arg builtin form:
//   (uint32x4 g0, int32x8 g1, int32x4 g2, int32x4 g3, int32x8 g4, int cpol)
static __device__ inline void tdm_load_tile_bf16(const void* gaddr, unsigned lds_off,
                                                 unsigned tensor_d0, unsigned tensor_d1,
                                                 unsigned tile_d0, unsigned tile_d1,
                                                 unsigned long long stride_elems,
                                                 unsigned pad_interval, unsigned pad_amount,
                                                 unsigned pad_enable) {
  unsigned long long ga = (unsigned long long)gaddr;
  u32x4 g0;
  g0[0] = 1u;                                        // count=1, user descriptor
  g0[1] = lds_off;                                   // lds byte address
  g0[2] = (unsigned)(ga & 0xffffffffu);              // global_addr[31:0]
  g0[3] = (unsigned)((ga >> 32) & 0x01ffffffu)       // global_addr[56:32]
        | (2u << 30);                                // type = 2 (image)
  i32x8 g1;
  g1[0] = (int)((1u << 16)                           // data_size = 1 (2 bytes)
        | (pad_enable << 20)
        | ((pad_interval & 7u) << 22)
        | ((pad_amount & 127u) << 25));
  g1[1] = (int)((tensor_d0 & 0xffffu) << 16);                                  // dim0[15:0]
  g1[2] = (int)(((tensor_d0 >> 16) & 0xffffu) | ((tensor_d1 & 0xffffu) << 16));// dim0[31:16], dim1[15:0]
  g1[3] = (int)(((tensor_d1 >> 16) & 0xffffu) | ((tile_d0 & 0xffffu) << 16));  // dim1[31:16], tile0
  g1[4] = (int)(tile_d1 & 0xffffu);                                            // tile1 (tile2=0)
  g1[5] = (int)(unsigned)(stride_elems & 0xffffffffu);                         // dim0_stride[31:0]
  g1[6] = (int)(unsigned)((stride_elems >> 32) & 0xffffu);                     // dim0_stride[47:32]
  g1[7] = 0;
  i32x4 z4 = {0, 0, 0, 0};
  i32x8 z8 = {0, 0, 0, 0, 0, 0, 0, 0};
  __builtin_amdgcn_tensor_load_to_lds(g0, g1, z4, z4, z8, 0);
}

// ---------------- fp32 -> bf16 convert ----------------
__global__ __launch_bounds__(256) void cvt_bf16_kernel(const float* __restrict__ src,
                                                       bf16* __restrict__ dst, int n) {
  int i = blockIdx.x * 256 + threadIdx.x;
  if (i < n) dst[i] = (bf16)src[i];
}

// ---------------- LayerNorm (D=512) -> bf16 ----------------
__global__ __launch_bounds__(256) void ln_bf16_kernel(const float* __restrict__ x,
                                                      const float* __restrict__ g,
                                                      const float* __restrict__ beta,
                                                      bf16* __restrict__ out) {
  int row = blockIdx.x;
  int tid = threadIdx.x;
  const float* xr = x + (size_t)row * cD;
  float v0 = xr[tid], v1 = xr[tid + 256];
  __shared__ float red[256];
  red[tid] = v0 + v1;
  __syncthreads();
  for (int off = 128; off > 0; off >>= 1) {
    if (tid < off) red[tid] += red[tid + off];
    __syncthreads();
  }
  float mean = red[0] * (1.0f / cD);
  __syncthreads();
  float d0 = v0 - mean, d1 = v1 - mean;
  red[tid] = d0 * d0 + d1 * d1;
  __syncthreads();
  for (int off = 128; off > 0; off >>= 1) {
    if (tid < off) red[tid] += red[tid + off];
    __syncthreads();
  }
  float rstd = rsqrtf(red[0] * (1.0f / cD) + 1e-5f);
  out[(size_t)row * cD + tid]       = (bf16)(d0 * rstd * g[tid] + beta[tid]);
  out[(size_t)row * cD + tid + 256] = (bf16)(d1 * rstd * g[tid + 256] + beta[tid + 256]);
}

// ---------------- WMMA GEMM: C[M,N] = A[M,K] @ B[K,N] (+epilogue) ----------------
// A tile DMA'd by the Tensor Data Mover (with LDS row padding to ld=40);
// B tile staged transposed by the threads so fragment reads are b128 vector loads.
// EPI 0: plain bf16 store
// EPI 1: fp32 store of acc + bias[n] + resid[m*N+n]
// EPI 2: bf16 store of gelu(acc + bias[n]) (exact erf gelu)
#define GBM 128
#define GBN 64
#define GBK 32
#define ALD 40   // padded LDS leading dim for A (32 data + 8 pad bf16 = 80B rows)
#define BLD 40   // padded LDS leading dim for B^T
template <int EPI>
__global__ __launch_bounds__(256) void gemm_wmma_kernel(const bf16* __restrict__ A,
                                                        const bf16* __restrict__ Bm,
                                                        const float* __restrict__ bias,
                                                        const float* __restrict__ resid,
                                                        void* __restrict__ outv,
                                                        int M, int N, int K) {
  __shared__ __align__(16) bf16 As[GBM][ALD];     // row-major, TDM-filled
  __shared__ __align__(16) bf16 Bt[GBN][BLD];     // transposed: Bt[n][k]
  int tid  = threadIdx.x;
  int wid  = tid >> 5;
  int lane = tid & 31;
  int wm = wid & 3;   // 4 row groups of 32
  int wn = wid >> 2;  // 2 col groups of 32
  int m0 = blockIdx.y * GBM;
  int n0 = blockIdx.x * GBN;

  C8 c[4];
#pragma unroll
  for (int f = 0; f < 4; ++f)
#pragma unroll
    for (int e = 0; e < 8; ++e) c[f].f[e] = 0.0f;

  for (int k0 = 0; k0 < K; k0 += GBK) {
    __syncthreads();  // protect LDS tiles from previous iteration's readers
    if (wid == 0) {
      // DMA the 128x32 A tile into LDS; pad 4 DWORDs after every 16 DWORDs
      // (i.e., +16B after each 64B row) -> effective LDS row stride = 80B = ALD.
      tdm_load_tile_bf16(&A[(size_t)m0 * K + k0], (unsigned)(size_t)&As[0][0],
                         (unsigned)K, (unsigned)M, GBK, GBM,
                         (unsigned long long)K,
                         /*pad_interval=*/3, /*pad_amount=*/3, /*pad_enable=*/1);
      __builtin_amdgcn_s_wait_tensorcnt(0);
    }
    // stage B tile transposed: Bt[n][k] = B[k0+k][n0+n]
#pragma unroll
    for (int idx = tid; idx < GBK * GBN; idx += 256) {
      int r = idx >> 6, cc = idx & 63;
      Bt[cc][r] = Bm[(size_t)(k0 + r) * N + n0 + cc];
    }
    __syncthreads();
    v16bf a0 = load_a_frag(&As[wm * 32][0],       ALD, lane);
    v16bf a1 = load_a_frag(&As[wm * 32 + 16][0],  ALD, lane);
    v16bf b0 = load_bT_frag(&Bt[wn * 32][0],      BLD, lane);
    v16bf b1 = load_bT_frag(&Bt[wn * 32 + 16][0], BLD, lane);
    c[0].v = wmma_bf16(a0, b0, c[0].v);
    c[1].v = wmma_bf16(a0, b1, c[1].v);
    c[2].v = wmma_bf16(a1, b0, c[2].v);
    c[3].v = wmma_bf16(a1, b1, c[3].v);
  }

  int kh = lane >> 4, nn = lane & 15;
#pragma unroll
  for (int f = 0; f < 4; ++f) {
    int ta = f >> 1, tb = f & 1;
#pragma unroll
    for (int e = 0; e < 8; ++e) {
      int m = m0 + wm * 32 + ta * 16 + e + 8 * kh;
      int n = n0 + wn * 32 + tb * 16 + nn;
      float val = c[f].f[e];
      if constexpr (EPI == 0) {
        ((bf16*)outv)[(size_t)m * N + n] = (bf16)val;
      } else if constexpr (EPI == 1) {
        val += bias[n] + resid[(size_t)m * N + n];
        ((float*)outv)[(size_t)m * N + n] = val;
      } else {
        val += bias[n];
        val = 0.5f * val * (1.0f + erff(val * 0.70710678118654752f));
        ((bf16*)outv)[(size_t)m * N + n] = (bf16)val;
      }
    }
  }
}

// ---------------- qrel[bh, i, r] = scale * dot(q_row, rel_k_emb[r]) ----------------
__global__ __launch_bounds__(256) void qrel_kernel(const bf16* __restrict__ q,
                                                   const float* __restrict__ relk,
                                                   float* __restrict__ qrel) {
  int idx = blockIdx.x * 256 + threadIdx.x;
  int total = cB * cH * cS * cNB;
  if (idx >= total) return;
  int r  = idx % cNB;
  int t  = idx / cNB;
  int i  = t % cS;
  int bh = t / cS;
  int b = bh >> 3, h = bh & 7;
  const bf16*  qp = q + ((size_t)(b * cS + i)) * cD + h * cDK;
  const float* rp = relk + (size_t)r * cDK;
  float acc = 0.0f;
#pragma unroll 16
  for (int d = 0; d < cDK; ++d) acc += (float)qp[d] * rp[d];
  qrel[idx] = acc * cSCALE;
}

// ---------------- flash attention with Shaw rel-pos (WMMA core) ----------------
// grid: (S/64, B*H), block 128 (4 waves); each wave owns 16 query rows.
// K/V chunks staged cooperatively in LDS (shared by all 4 waves); V transposed
// so P@V B-fragments are contiguous b128 LDS reads.
#define KLD 72   // Ks row stride (64 data + 8 pad; 144B rows)
#define VLD 40   // Vt row stride (32 data + 8 pad; 80B rows)
#define PLD 40   // Pstage row stride
__global__ __launch_bounds__(128) void attn_kernel(const bf16* __restrict__ q,
                                                   const bf16* __restrict__ k,
                                                   const bf16* __restrict__ v,
                                                   const float* __restrict__ qrel,
                                                   const float* __restrict__ relv,
                                                   bf16* __restrict__ ctx) {
  __shared__ __align__(16) bf16 Ks[32][KLD];          // Ks[key][d]
  __shared__ __align__(16) bf16 Vt[64][VLD];          // Vt[d][key]
  __shared__ __align__(16) bf16 Pstage[4][16][PLD];
  __shared__ float bucketS[4][16][cNB + 1];
  __shared__ float alphaS[4][16];

  int tid  = threadIdx.x;
  int w    = tid >> 5;
  int lane = tid & 31;
  int kh   = lane >> 4;
  int ln   = lane & 15;
  int bh   = blockIdx.y;
  int b = bh >> 3, h = bh & 7;
  int i0 = blockIdx.x * 64 + w * 16;

  const bf16*  qbase   = q + ((size_t)(b * cS + i0)) * cD + h * cDK;
  const bf16*  kg      = k + ((size_t)(b * cS)) * cD + h * cDK;
  const bf16*  vg      = v + ((size_t)(b * cS)) * cD + h * cDK;
  const float* qrelrow = qrel + ((size_t)bh * cS) * cNB;

  for (int idx = tid; idx < 4 * 16 * (cNB + 1); idx += 128)
    ((float*)bucketS)[idx] = 0.0f;

  v16bf qa0 = load_a_frag(qbase,      cD, lane);   // d = 0..31
  v16bf qa1 = load_a_frag(qbase + 32, cD, lane);   // d = 32..63

  C8 o[4];
#pragma unroll
  for (int f = 0; f < 4; ++f)
#pragma unroll
    for (int e = 0; e < 8; ++e) o[f].f[e] = 0.0f;

  float mrow[8], lrow[8];
#pragma unroll
  for (int e = 0; e < 8; ++e) { mrow[e] = -1e30f; lrow[e] = 0.0f; }

  for (int kc = 0; kc < cS; kc += 32) {
    __syncthreads();  // previous iteration's LDS readers done
    // --- stage K (row-major) and V (transposed) chunk: 32 keys x 64 dims ---
#pragma unroll
    for (int idx = tid; idx < 32 * 64; idx += 128) {
      int r = idx >> 6, cc = idx & 63;
      bf16 kvK = kg[(size_t)(kc + r) * cD + cc];
      bf16 kvV = vg[(size_t)(kc + r) * cD + cc];
      Ks[r][cc] = kvK;
      Vt[cc][r] = kvV;
    }
    __syncthreads();

    float sc[2][8];
    // --- scores: two 16x16 tiles via WMMA (K^T frags = b128 LDS reads) ---
#pragma unroll
    for (int t = 0; t < 2; ++t) {
      v16bf kb0 = load_bT_frag(&Ks[t * 16][0],      KLD, lane);  // d 0..31
      v16bf kb1 = load_bT_frag(&Ks[t * 16][0] + 32, KLD, lane);  // d 32..63
      C8 cf;
#pragma unroll
      for (int e = 0; e < 8; ++e) cf.f[e] = 0.0f;
      cf.v = wmma_bf16(qa0, kb0, cf.v);
      cf.v = wmma_bf16(qa1, kb1, cf.v);
      int j = kc + t * 16 + ln;
#pragma unroll
      for (int e = 0; e < 8; ++e) {
        int i = i0 + e + 8 * kh;
        int delta = iclamp(j - i, -32, 32) + 32;
        sc[t][e] = cf.f[e] * cSCALE + qrelrow[(size_t)i * cNB + delta];
      }
    }
    // --- online softmax: per-row max/sum across 16 lanes ---
    float alpha[8], mnew[8];
#pragma unroll
    for (int e = 0; e < 8; ++e) {
      float cm = fmaxf(sc[0][e], sc[1][e]);
      cm = fmaxf(cm, __shfl_xor(cm, 1, 32));
      cm = fmaxf(cm, __shfl_xor(cm, 2, 32));
      cm = fmaxf(cm, __shfl_xor(cm, 4, 32));
      cm = fmaxf(cm, __shfl_xor(cm, 8, 32));
      mnew[e]  = fmaxf(mrow[e], cm);
      alpha[e] = __expf(mrow[e] - mnew[e]);
      mrow[e]  = mnew[e];
    }
    if (ln == 0) {
#pragma unroll
      for (int e = 0; e < 8; ++e) alphaS[w][kh * 8 + e] = alpha[e];
    }
    __syncthreads();
#pragma unroll
    for (int e = 0; e < 8; ++e) {
      float p0 = __expf(sc[0][e] - mnew[e]);
      float p1 = __expf(sc[1][e] - mnew[e]);
      sc[0][e] = p0; sc[1][e] = p1;
      float rs = p0 + p1;
      rs += __shfl_xor(rs, 1, 32);
      rs += __shfl_xor(rs, 2, 32);
      rs += __shfl_xor(rs, 4, 32);
      rs += __shfl_xor(rs, 8, 32);
      lrow[e] = lrow[e] * alpha[e] + rs;
    }
#pragma unroll
    for (int f = 0; f < 4; ++f)
#pragma unroll
      for (int e = 0; e < 8; ++e) o[f].f[e] *= alpha[e];
    // rescale bucket sums by alpha (rel_v accumulator)
    for (int idx = lane; idx < 16 * cNB; idx += 32) {
      int row = idx / cNB, nb = idx % cNB;
      bucketS[w][row][nb] *= alphaS[w][row];
    }
    __syncthreads();
    // accumulate buckets + stage P in bf16 for the P@V WMMA
#pragma unroll
    for (int t = 0; t < 2; ++t) {
#pragma unroll
      for (int e = 0; e < 8; ++e) {
        int row = e + 8 * kh;
        int j = kc + t * 16 + ln;
        int i = i0 + row;
        int delta = iclamp(j - i, -32, 32) + 32;
        atomicAdd(&bucketS[w][row][delta], sc[t][e]);
        Pstage[w][row][t * 16 + ln] = (bf16)sc[t][e];
      }
    }
    __syncthreads();
    // --- ctx += P(16x32) @ V(32x64): 4 WMMAs, V frags from transposed LDS ---
    v16bf pa = load_a_frag(&Pstage[w][0][0], PLD, lane);
#pragma unroll
    for (int dt = 0; dt < 4; ++dt) {
      v16bf vb = load_bT_frag(&Vt[dt * 16][0], VLD, lane);
      o[dt].v = wmma_bf16(pa, vb, o[dt].v);
    }
  }
  __syncthreads();  // buckets final

  // --- finalize: (O + bucket@rel_v) / l ---
  float invl[8];
#pragma unroll
  for (int e = 0; e < 8; ++e) invl[e] = 1.0f / lrow[e];
#pragma unroll
  for (int dt = 0; dt < 4; ++dt) {
#pragma unroll
    for (int e = 0; e < 8; ++e) {
      int row = e + 8 * kh;
      int d = dt * 16 + ln;
      float acc = 0.0f;
      for (int nb = 0; nb < cNB; ++nb)
        acc += bucketS[w][row][nb] * relv[(size_t)nb * cDK + d];
      float val = (o[dt].f[e] + acc) * invl[e];
      int i = i0 + row;
      ctx[((size_t)(b * cS + i)) * cD + h * cDK + d] = (bf16)val;
    }
  }
}

// ---------------- host orchestration ----------------
extern "C" void kernel_launch(void* const* d_in, const int* in_sizes, int n_in,
                              void* d_out, int out_size, void* d_ws, size_t ws_size,
                              hipStream_t stream) {
  const float* x     = (const float*)d_in[0];
  const float* w_q   = (const float*)d_in[1];
  const float* w_k   = (const float*)d_in[2];
  const float* w_v   = (const float*)d_in[3];
  const float* w_o   = (const float*)d_in[4];
  const float* b_o   = (const float*)d_in[5];
  const float* rel_k = (const float*)d_in[6];
  const float* rel_v = (const float*)d_in[7];
  const float* w1    = (const float*)d_in[8];
  const float* b1    = (const float*)d_in[9];
  const float* w2    = (const float*)d_in[10];
  const float* b2    = (const float*)d_in[11];
  const float* g1    = (const float*)d_in[12];
  const float* be1   = (const float*)d_in[13];
  const float* g2    = (const float*)d_in[14];
  const float* be2   = (const float*)d_in[15];
  float* out = (float*)d_out;

  char* ws = (char*)d_ws;
  auto take = [&](size_t bytes) -> char* {
    char* p = ws;
    ws += (bytes + 255) & ~(size_t)255;
    return p;
  };
  const size_t M = (size_t)cB * cS;  // 4096 token rows

  bf16* wq_b  = (bf16*)take((size_t)cD * cD * 2);
  bf16* wk_b  = (bf16*)take((size_t)cD * cD * 2);
  bf16* wv_b  = (bf16*)take((size_t)cD * cD * 2);
  bf16* wo_b  = (bf16*)take((size_t)cD * cD * 2);
  bf16* w1_b  = (bf16*)take((size_t)cD * cFF * 2);
  bf16* w2_b  = (bf16*)take((size_t)cFF * cD * 2);
  bf16* hn1   = (bf16*)take(M * cD * 2);
  bf16* qb    = (bf16*)take(M * cD * 2);
  bf16* kb    = (bf16*)take(M * cD * 2);
  bf16* vb    = (bf16*)take(M * cD * 2);
  float* qrel = (float*)take((size_t)cB * cH * cS * cNB * 4);
  bf16* ctx   = (bf16*)take(M * cD * 2);
  float* x2   = (float*)take(M * cD * 4);
  bf16* hn2   = (bf16*)take(M * cD * 2);
  bf16* ff1   = (bf16*)take(M * cFF * 2);

  // 1. weight conversion
  {
    int n = cD * cD;
    cvt_bf16_kernel<<<(n + 255) / 256, 256, 0, stream>>>(w_q, wq_b, n);
    cvt_bf16_kernel<<<(n + 255) / 256, 256, 0, stream>>>(w_k, wk_b, n);
    cvt_bf16_kernel<<<(n + 255) / 256, 256, 0, stream>>>(w_v, wv_b, n);
    cvt_bf16_kernel<<<(n + 255) / 256, 256, 0, stream>>>(w_o, wo_b, n);
    int n2 = cD * cFF;
    cvt_bf16_kernel<<<(n2 + 255) / 256, 256, 0, stream>>>(w1, w1_b, n2);
    cvt_bf16_kernel<<<(n2 + 255) / 256, 256, 0, stream>>>(w2, w2_b, n2);
  }
  // 2. LN1
  ln_bf16_kernel<<<(int)M, 256, 0, stream>>>(x, g1, be1, hn1);
  // 3. QKV projections
  {
    dim3 g(cD / GBN, (int)(M / GBM));
    gemm_wmma_kernel<0><<<g, 256, 0, stream>>>(hn1, wq_b, nullptr, nullptr, qb, (int)M, cD, cD);
    gemm_wmma_kernel<0><<<g, 256, 0, stream>>>(hn1, wk_b, nullptr, nullptr, kb, (int)M, cD, cD);
    gemm_wmma_kernel<0><<<g, 256, 0, stream>>>(hn1, wv_b, nullptr, nullptr, vb, (int)M, cD, cD);
  }
  // 4. q . rel_k_emb (65 buckets per row, scaled)
  {
    int total = cB * cH * cS * cNB;
    qrel_kernel<<<(total + 255) / 256, 256, 0, stream>>>(qb, rel_k, qrel);
  }
  // 5. flash attention + rel-pos bias + rel_v bucket term
  attn_kernel<<<dim3(cS / 64, cB * cH), 128, 0, stream>>>(qb, kb, vb, qrel, rel_v, ctx);
  // 6. output projection + bias + residual -> x2 (fp32)
  gemm_wmma_kernel<1><<<dim3(cD / GBN, (int)(M / GBM)), 256, 0, stream>>>(
      ctx, wo_b, b_o, x, x2, (int)M, cD, cD);
  // 7. LN2
  ln_bf16_kernel<<<(int)M, 256, 0, stream>>>(x2, g2, be2, hn2);
  // 8. FFN up + exact GELU
  gemm_wmma_kernel<2><<<dim3(cFF / GBN, (int)(M / GBM)), 256, 0, stream>>>(
      hn2, w1_b, b1, nullptr, ff1, (int)M, cFF, cD);
  // 9. FFN down + bias + residual -> out
  gemm_wmma_kernel<1><<<dim3(cD / GBN, (int)(M / GBM)), 256, 0, stream>>>(
      ff1, w2_b, b2, x2, out, (int)M, cD, cFF);
}
